// GraphSAGE_4569845203115
// MI455X (gfx1250) — compile-verified, matrix operand order checked by reading, not run on hardware
//
#include <hip/hip_runtime.h>
#include <hip/hip_bf16.h>

typedef __attribute__((ext_vector_type(16))) _Float16 v16h;
typedef __attribute__((ext_vector_type(8)))  float    v8f;
typedef __attribute__((ext_vector_type(4)))  float    v4f;

#define IN_DIM 128
#define HID    64
#define NOUT   3
#define WPB    8          // waves per block

// ---------------------------------------------------------------- utilities

__global__ void zero_f32(float* __restrict__ p, size_t n) {
  size_t i = (size_t)blockIdx.x * blockDim.x + threadIdx.x;
  if (i < n) p[i] = 0.0f;
}

// one thread per edge: in-degree count (as float for later reciprocal)
__global__ void degree_count(const int* __restrict__ ei, float* __restrict__ cnt, int E) {
  int e = blockIdx.x * blockDim.x + threadIdx.x;
  if (e < E) atomicAdd(cnt + ei[E + e], 1.0f);
}

// 32 lanes per edge, float4 per lane: agg1[dst] += x[src]  (128 floats/edge)
__global__ void scatter_add_128(const float* __restrict__ x, const int* __restrict__ ei,
                                float* __restrict__ agg, int E) {
  long long t = (long long)blockIdx.x * blockDim.x + threadIdx.x;
  int e = (int)(t >> 5);
  if (e >= E) return;
  int c = ((int)t & 31) * 4;
  int s = ei[e], d = ei[E + e];
  v4f v = *(const v4f*)(x + (size_t)s * IN_DIM + c);
  float* out = agg + (size_t)d * IN_DIM + c;
  atomicAdd(out + 0, v[0]); atomicAdd(out + 1, v[1]);
  atomicAdd(out + 2, v[2]); atomicAdd(out + 3, v[3]);
}

// 16 lanes per edge: agg2[dst] += h[src]  (64 floats/edge)
__global__ void scatter_add_64(const float* __restrict__ h, const int* __restrict__ ei,
                               float* __restrict__ agg, int E) {
  long long t = (long long)blockIdx.x * blockDim.x + threadIdx.x;
  int e = (int)(t >> 4);
  if (e >= E) return;
  int c = ((int)t & 15) * 4;
  int s = ei[e], d = ei[E + e];
  v4f v = *(const v4f*)(h + (size_t)s * HID + c);
  float* out = agg + (size_t)d * HID + c;
  atomicAdd(out + 0, v[0]); atomicAdd(out + 1, v[1]);
  atomicAdd(out + 2, v[2]); atomicAdd(out + 3, v[3]);
}

// ------------------------------------------------- WMMA fragment assembly
// ISA 7.12.2, 16-bit A-matrix 16x32 (MxK), wave32:
//   lanes 0-15 : M=lane,    elems = K {k0+0..7}  then {k0+16..23}
//   lanes 16-31: M=lane-16, elems = K {k0+8..15} then {k0+24..31}
__device__ __forceinline__ v16h load_a_frag_f32(const float* __restrict__ rowBase,
                                                int k0, float scale, int khalf) {
  int g0 = khalf * 8;
  int g1 = 16 + khalf * 8;
  v4f a0 = *(const v4f*)(rowBase + k0 + g0);
  v4f a1 = *(const v4f*)(rowBase + k0 + g0 + 4);
  v4f a2 = *(const v4f*)(rowBase + k0 + g1);
  v4f a3 = *(const v4f*)(rowBase + k0 + g1 + 4);
  v16h r;
#pragma unroll
  for (int i = 0; i < 4; ++i) {
    r[i]      = (_Float16)(a0[i] * scale);
    r[4 + i]  = (_Float16)(a1[i] * scale);
    r[8 + i]  = (_Float16)(a2[i] * scale);
    r[12 + i] = (_Float16)(a3[i] * scale);
  }
  return r;
}

// h = relu( (agg1/deg) @ Wl1 + x @ Wr1 + b1 )
// 256 threads = 8 waves; each wave produces one 16-row strip across all 4
// column tiles of HID=64 (A fragments reused 4x, 32 WMMAs per wave).
//
// Weights are pre-packed in LDS in *fragment-major* order so a lane's 16
// f16 B-elements are one contiguous 32-byte chunk (-> 2x ds_load_b128):
//   group g = (colTile*2 + khalf)*4 + kChunk,  within group: [n*16 + j]
//   element j maps to K = kChunk*32 + khalf*8 + j + (j>=8 ? 8 : 0)
__global__ void __launch_bounds__(256)
sage_layer1_wmma(const float* __restrict__ x, const float* __restrict__ agg,
                 const float* __restrict__ cnt,
                 const float* __restrict__ Wl, const float* __restrict__ Wr,
                 const float* __restrict__ bias,
                 float* __restrict__ h, int N, int rowTiles) {
  __shared__ _Float16 sW[2][32][256];    // [mat][group][n*16+j], 32 KB total

  for (int idx = threadIdx.x; idx < 32 * 256; idx += 256) {
    int g  = idx >> 8;
    int n  = (idx >> 4) & 15;
    int j  = idx & 15;
    int ct = g >> 3;
    int kh = (g >> 2) & 1;
    int kc = g & 3;
    int k  = kc * 32 + kh * 8 + j + ((j >= 8) ? 8 : 0);
    int col = ct * 16 + n;
    sW[0][g][n * 16 + j] = (_Float16)Wl[k * HID + col];
    sW[1][g][n * 16 + j] = (_Float16)Wr[k * HID + col];
  }
  __syncthreads();

  int wave = threadIdx.x >> 5;
  int lid  = threadIdx.x & 31;
  int rowTile = blockIdx.x * WPB + wave;
  if (rowTile >= rowTiles) return;       // wave-uniform: EXEC stays all-ones

  int r0    = rowTile * 16;
  int m     = lid & 15;
  int khalf = lid >> 4;
  int row = r0 + m; if (row >= N) row = N - 1;   // clamp keeps full-wave loads

  float inv = 1.0f / fmaxf(cnt[row], 1.0f);      // fold mean into A-load
  const float* aggRow = agg + (size_t)row * IN_DIM;
  const float* xRow   = x   + (size_t)row * IN_DIM;

  v8f acc[4] = {{}, {}, {}, {}};
#pragma unroll
  for (int kc = 0; kc < 4; ++kc) {
    int k0 = kc * 32;
    v16h aA = load_a_frag_f32(aggRow, k0, inv,  khalf);
    v16h aX = load_a_frag_f32(xRow,   k0, 1.0f, khalf);
#pragma unroll
    for (int ct = 0; ct < 4; ++ct) {
      int g = (ct * 2 + khalf) * 4 + kc;
      v16h bL = *(const v16h*)&sW[0][g][m * 16];   // 32B contiguous -> b128 x2
      v16h bR = *(const v16h*)&sW[1][g][m * 16];
      acc[ct] = __builtin_amdgcn_wmma_f32_16x16x32_f16(false, aA, false, bL,
                                                       (short)0, acc[ct],
                                                       false, false);
      acc[ct] = __builtin_amdgcn_wmma_f32_16x16x32_f16(false, aX, false, bR,
                                                       (short)0, acc[ct],
                                                       false, false);
    }
  }

  // C/D layout: VGPR r -> M = r + 8*khalf, N = lane&15 (per column tile)
#pragma unroll
  for (int ct = 0; ct < 4; ++ct) {
    int n = ct * 16 + m;
    float bv = bias[n];
#pragma unroll
    for (int r = 0; r < 8; ++r) {
      int orow = r0 + r + 8 * khalf;
      if (orow < N) h[(size_t)orow * HID + n] = fmaxf(acc[ct][r] + bv, 0.0f);
    }
  }
}

// out = log_softmax( (agg2/deg) @ Wl2 + h @ Wr2 + b2 ), OUT_DIM = 3
__global__ void sage_layer2(const float* __restrict__ h, const float* __restrict__ agg2,
                            const float* __restrict__ cnt,
                            const float* __restrict__ Wl, const float* __restrict__ Wr,
                            const float* __restrict__ bias,
                            float* __restrict__ out, int N) {
  int nidx = blockIdx.x * blockDim.x + threadIdx.x;
  if (nidx >= N) return;
  float inv = 1.0f / fmaxf(cnt[nidx], 1.0f);
  const float* hr = h    + (size_t)nidx * HID;
  const float* ar = agg2 + (size_t)nidx * HID;
  float o0 = bias[0], o1 = bias[1], o2 = bias[2];
#pragma unroll 8
  for (int k = 0; k < HID; ++k) {
    float av = ar[k] * inv;
    float hv = hr[k];
    o0 += av * Wl[k * NOUT + 0] + hv * Wr[k * NOUT + 0];
    o1 += av * Wl[k * NOUT + 1] + hv * Wr[k * NOUT + 1];
    o2 += av * Wl[k * NOUT + 2] + hv * Wr[k * NOUT + 2];
  }
  float mx  = fmaxf(o0, fmaxf(o1, o2));
  float lse = logf(expf(o0 - mx) + expf(o1 - mx) + expf(o2 - mx));
  out[(size_t)nidx * NOUT + 0] = o0 - mx - lse;
  out[(size_t)nidx * NOUT + 1] = o1 - mx - lse;
  out[(size_t)nidx * NOUT + 2] = o2 - mx - lse;
}

// ----------------------------------------------------------------- launch

extern "C" void kernel_launch(void* const* d_in, const int* in_sizes, int n_in,
                              void* d_out, int out_size, void* d_ws, size_t ws_size,
                              hipStream_t stream) {
  (void)n_in; (void)out_size; (void)ws_size;
  const float* x   = (const float*)d_in[0];
  const int*   ei  = (const int*)d_in[1];   // [2][E]: row 0 = src, row 1 = dst
  const float* Wl1 = (const float*)d_in[2];
  const float* Wr1 = (const float*)d_in[3];
  const float* b1  = (const float*)d_in[4];
  const float* Wl2 = (const float*)d_in[5];
  const float* Wr2 = (const float*)d_in[6];
  const float* b2  = (const float*)d_in[7];

  int N = in_sizes[0] / IN_DIM;
  int E = in_sizes[1] / 2;

  // workspace layout (floats): agg1[N*128] | agg2[N*64] | h[N*64] | cnt[N]
  float* agg1 = (float*)d_ws;
  float* agg2 = agg1 + (size_t)N * IN_DIM;
  float* hbuf = agg2 + (size_t)N * HID;
  float* cnt  = hbuf + (size_t)N * HID;

  size_t nz = (size_t)N * (IN_DIM + HID);         // agg1 + agg2 contiguous
  zero_f32<<<(int)((nz + 255) / 256), 256, 0, stream>>>(agg1, nz);
  zero_f32<<<(N + 255) / 256, 256, 0, stream>>>(cnt, (size_t)N);

  degree_count<<<(E + 255) / 256, 256, 0, stream>>>(ei, cnt, E);

  long long t1 = (long long)E * 32;
  scatter_add_128<<<(int)((t1 + 255) / 256), 256, 0, stream>>>(x, ei, agg1, E);

  int rowTiles = (N + 15) / 16;
  int blocks   = (rowTiles + WPB - 1) / WPB;
  sage_layer1_wmma<<<blocks, 256, 0, stream>>>(x, agg1, cnt, Wl1, Wr1, b1,
                                               hbuf, N, rowTiles);

  long long t2 = (long long)E * 16;
  scatter_add_64<<<(int)((t2 + 255) / 256), 256, 0, stream>>>(hbuf, ei, agg2, E);

  sage_layer2<<<(N + 255) / 256, 256, 0, stream>>>(hbuf, agg2, cnt, Wl2, Wr2, b2,
                                                   (float*)d_out, N);
}
